// LuongAttention_14259291423020
// MI455X (gfx1250) — matrix operand order repaired and checked by reading, not compiled, not run on hardware
//
#include <hip/hip_runtime.h>
#include <stdint.h>

#define ENC_L 2048
#define DEC_L 512
#define NBAT  32
#define HDIM  256
#define NEGVAL (-1.0e12f)

typedef __attribute__((ext_vector_type(16))) __bf16 v16bf;
typedef __attribute__((ext_vector_type(8)))  float  v8f;

union BfFrag {
    v16bf v;
    uint4 q[2];
    unsigned short u[16];
};

__device__ __forceinline__ unsigned short f2bf(float f) {
    unsigned int u = __float_as_uint(f);
    u += 0x7FFFu + ((u >> 16) & 1u);   // round-to-nearest-even
    return (unsigned short)(u >> 16);
}

__device__ __forceinline__ v8f wmma_bf16(const BfFrag& a, const BfFrag& b, v8f c) {
    // D = A(16x32 bf16) x B(32x16 bf16) + C(16x16 f32)
    return __builtin_amdgcn_wmma_f32_16x16x32_bf16(
        false, a.v, false, b.v, (short)0, c, false, false);
}

// ---------------------------------------------------------------------------
// Convert encoder (S,B,H) f32 -> encB [b][s][h] bf16 and encT [b][h][s] bf16.
// One block per (b, 64-row s-tile, 64-col h-tile); LDS transpose for encT.
// ---------------------------------------------------------------------------
__global__ void k_conv_enc(const float* __restrict__ enc,
                           unsigned short* __restrict__ encB,
                           unsigned short* __restrict__ encT) {
    __shared__ unsigned short tile[64][65];
    const int gid  = blockIdx.x;          // 32 * 32 * 4 = 4096 blocks
    const int b    = gid >> 7;
    const int rem  = gid & 127;
    const int sblk = rem >> 2;            // 0..31 (64 rows each)
    const int hblk = rem & 3;             // 0..3  (64 cols each)
    const int t    = threadIdx.x;         // 256 threads

    for (int i = 0; i < 16; ++i) {
        const int lin = t + 256 * i;      // 4096 elements
        const int sl  = lin >> 6;
        const int hl  = lin & 63;
        const int s   = sblk * 64 + sl;
        const int h   = hblk * 64 + hl;
        const float f = enc[((size_t)s * NBAT + b) * HDIM + h];
        const unsigned short v = f2bf(f);
        encB[((size_t)b * ENC_L + s) * HDIM + h] = v;
        tile[sl][hl] = v;
    }
    __syncthreads();
    for (int i = 0; i < 16; ++i) {
        const int lin = t + 256 * i;
        const int hl  = lin >> 6;
        const int sl  = lin & 63;
        encT[((size_t)b * HDIM + hblk * 64 + hl) * ENC_L + sblk * 64 + sl] =
            tile[sl][hl];
    }
}

// decoder (D,B,H) f32 -> decB [b][d][h] bf16
__global__ void k_conv_dec(const float* __restrict__ dec,
                           unsigned short* __restrict__ decB) {
    const size_t tid = (size_t)blockIdx.x * 256 + threadIdx.x;
    const int h = (int)(tid & 255);
    const int b = (int)((tid >> 8) & 31);
    const int d = (int)(tid >> 13);
    decB[((size_t)b * DEC_L + d) * HDIM + h] =
        f2bf(dec[((size_t)d * NBAT + b) * HDIM + h]);
}

// W (Hd,He) f32 -> wT [he][hd] bf16 (transposed so K is contiguous per row)
__global__ void k_conv_w(const float* __restrict__ W,
                         unsigned short* __restrict__ wT) {
    const int tid = blockIdx.x * 256 + threadIdx.x;   // 65536
    const int hd  = tid & 255;
    const int he  = tid >> 8;
    wT[(size_t)he * HDIM + hd] = f2bf(W[(size_t)hd * HDIM + he]);
}

// ---------------------------------------------------------------------------
// proj[b][d][he] = sum_hd dec[b][d][hd] * W[hd][he]   (bf16 in, bf16 out)
// One wave per 16x16 tile, K = 256 (8 WMMAs). 8 waves / block.
// ---------------------------------------------------------------------------
__global__ __launch_bounds__(256) void k_proj(
        const unsigned short* __restrict__ decB,
        const unsigned short* __restrict__ wT,
        unsigned short* __restrict__ projB) {
    const int lane = threadIdx.x & 31;
    const int wave = threadIdx.x >> 5;
    const int tile = blockIdx.x * 8 + wave;   // 16384 tiles
    const int b    = tile >> 9;
    const int rem  = tile & 511;
    const int dt   = rem >> 4;
    const int ht   = rem & 15;

    const unsigned short* arow =
        decB + ((size_t)b * DEC_L + dt * 16 + (lane & 15)) * HDIM;
    const unsigned short* brow = wT + (size_t)(ht * 16 + (lane & 15)) * HDIM;
    const int ahi = 8 * (lane >> 4);
    const int bhi = 8 * (lane >> 4);

    v8f acc = {};
    for (int kc = 0; kc < 8; ++kc) {
        BfFrag a, bb;
        const int h0 = kc * 32 + ahi;
        a.q[0]  = *(const uint4*)(arow + h0);
        a.q[1]  = *(const uint4*)(arow + h0 + 16);
        const int k0 = kc * 32 + bhi;
        bb.q[0] = *(const uint4*)(brow + k0);
        bb.q[1] = *(const uint4*)(brow + k0 + 16);
        acc = wmma_bf16(a, bb, acc);
    }
    const int rbase = 8 * (lane >> 4);
    const int he    = ht * 16 + (lane & 15);
    for (int r = 0; r < 8; ++r)
        projB[((size_t)b * DEC_L + dt * 16 + rbase + r) * HDIM + he] =
            f2bf(acc[r]);
}

// ---------------------------------------------------------------------------
// Fused: score = proj @ enc^T (+mask) -> softmax -> weights out -> attn out.
// One block per (batch, 16-decoder-row strip). 512 threads = 16 waves.
// LDS holds the full 16x2048 f32 score strip (128 KB).
// ---------------------------------------------------------------------------
__global__ __launch_bounds__(512) void k_attn_fused(
        const unsigned short* __restrict__ projB,
        const unsigned short* __restrict__ encB,
        const unsigned short* __restrict__ encT,
        const int* __restrict__ mask,
        float* __restrict__ attnOut,     // (dec_len, B, He)
        float* __restrict__ wOut) {      // (B*dec_len, enc_len)
    __shared__ float sS[16 * ENC_L];     // 131072 bytes

    const int dt   = blockIdx.x & 31;
    const int b    = blockIdx.x >> 5;
    const int lane = threadIdx.x & 31;
    const int wave = threadIdx.x >> 5;
    const int lrow = lane & 15;
    const int ahi  = 8 * (lane >> 4);
    const int bhi  = 8 * (lane >> 4);

    // ---- Phase A: scores. Each wave owns 8 of the 128 s-tiles. ----
    {
        // Preload the shared A operand (proj strip, K=256) once per wave.
        const unsigned short* arow =
            projB + ((size_t)b * DEC_L + dt * 16 + lrow) * HDIM;
        BfFrag af[8];
        for (int kc = 0; kc < 8; ++kc) {
            const int h0 = kc * 32 + ahi;
            af[kc].q[0] = *(const uint4*)(arow + h0);
            af[kc].q[1] = *(const uint4*)(arow + h0 + 16);
        }
        for (int t8 = 0; t8 < 8; ++t8) {
            const int s0 = (wave * 8 + t8) * 16;
            const unsigned short* brow =
                encB + ((size_t)b * ENC_L + s0 + lrow) * HDIM;
            v8f acc = {};
            for (int kc = 0; kc < 8; ++kc) {
                BfFrag bb;
                const int k0 = kc * 32 + bhi;
                bb.q[0] = *(const uint4*)(brow + k0);
                bb.q[1] = *(const uint4*)(brow + k0 + 16);
                acc = wmma_bf16(af[kc], bb, acc);
            }
            const int col = s0 + lrow;
            const int msk = mask[(size_t)col * NBAT + b];
            for (int r = 0; r < 8; ++r)
                sS[(ahi + r) * ENC_L + col] = (msk == 0) ? NEGVAL : acc[r];
        }
    }
    __syncthreads();

    // ---- Phase B: softmax, one wave per row; write normalized weights. ----
    {
        float* rowp = sS + wave * ENC_L;
        float mx = -3.4e38f;
        for (int c = lane; c < ENC_L; c += 32) mx = fmaxf(mx, rowp[c]);
        for (int off = 16; off > 0; off >>= 1)
            mx = fmaxf(mx, __shfl_xor(mx, off));
        float sum = 0.0f;
        for (int c = lane; c < ENC_L; c += 32) {
            const float e = __expf(rowp[c] - mx);
            rowp[c] = e;
            sum += e;
        }
        for (int off = 16; off > 0; off >>= 1)
            sum += __shfl_xor(sum, off);
        const float inv = 1.0f / sum;
        float* wrow = wOut + ((size_t)b * DEC_L + dt * 16 + wave) * ENC_L;
        for (int c = lane; c < ENC_L; c += 32) {
            const float p = rowp[c] * inv;
            rowp[c] = p;
            __builtin_nontemporal_store(p, wrow + c);   // streaming (NT)
        }
    }
    __syncthreads();

    // ---- Phase C: attn = P(16x2048) @ enc(2048x256); one h-tile / wave. ----
    {
        const int hbase = wave * 16;
        const float* arowp = sS + lrow * ENC_L;
        const unsigned short* brow =
            encT + ((size_t)b * HDIM + hbase + lrow) * ENC_L;
        v8f acc = {};
        for (int kc = 0; kc < 64; ++kc) {
            BfFrag a, bb;
            const int sa = kc * 32 + ahi;
            const float4 f0 = *(const float4*)(arowp + sa);
            const float4 f1 = *(const float4*)(arowp + sa + 4);
            const float4 f2 = *(const float4*)(arowp + sa + 16);
            const float4 f3 = *(const float4*)(arowp + sa + 20);
            a.u[0]  = f2bf(f0.x); a.u[1]  = f2bf(f0.y);
            a.u[2]  = f2bf(f0.z); a.u[3]  = f2bf(f0.w);
            a.u[4]  = f2bf(f1.x); a.u[5]  = f2bf(f1.y);
            a.u[6]  = f2bf(f1.z); a.u[7]  = f2bf(f1.w);
            a.u[8]  = f2bf(f2.x); a.u[9]  = f2bf(f2.y);
            a.u[10] = f2bf(f2.z); a.u[11] = f2bf(f2.w);
            a.u[12] = f2bf(f3.x); a.u[13] = f2bf(f3.y);
            a.u[14] = f2bf(f3.z); a.u[15] = f2bf(f3.w);
            const int k0 = kc * 32 + bhi;
            bb.q[0] = *(const uint4*)(brow + k0);
            bb.q[1] = *(const uint4*)(brow + k0 + 16);
            acc = wmma_bf16(a, bb, acc);
        }
        const int h = hbase + lrow;
        for (int r = 0; r < 8; ++r) {
            const int dg = dt * 16 + ahi + r;
            attnOut[((size_t)dg * NBAT + b) * HDIM + h] = acc[r];
        }
    }
}

// ---------------------------------------------------------------------------
extern "C" void kernel_launch(void* const* d_in, const int* in_sizes, int n_in,
                              void* d_out, int out_size, void* d_ws, size_t ws_size,
                              hipStream_t stream) {
    (void)in_sizes; (void)n_in; (void)out_size; (void)ws_size;
    const float* enc  = (const float*)d_in[0];   // (2048,32,256)
    const float* dec  = (const float*)d_in[1];   // (512,32,256)
    const float* W    = (const float*)d_in[2];   // (256,256)
    const int*   mask = (const int*)d_in[3];     // (2048,32)

    float* attnOut = (float*)d_out;                              // 512*32*256
    float* wOut    = attnOut + (size_t)DEC_L * NBAT * HDIM;      // 32*512*2048

    // Workspace layout (bf16 elements), total ~84 MB.
    unsigned short* ws   = (unsigned short*)d_ws;
    const size_t encElems = (size_t)NBAT * ENC_L * HDIM;         // 16,777,216
    const size_t decElems = (size_t)NBAT * DEC_L * HDIM;         //  4,194,304
    unsigned short* encB  = ws;
    unsigned short* encT  = encB + encElems;
    unsigned short* decB  = encT + encElems;
    unsigned short* projB = decB + decElems;
    unsigned short* wT    = projB + decElems;

    k_conv_enc<<<4096, 256, 0, stream>>>(enc, encB, encT);
    k_conv_dec<<<(unsigned)(decElems / 256), 256, 0, stream>>>(dec, decB);
    k_conv_w  <<<256, 256, 0, stream>>>(W, wT);
    k_proj    <<<2048, 256, 0, stream>>>(decB, wT, projB);
    k_attn_fused<<<NBAT * (DEC_L / 16), 512, 0, stream>>>(
        projB, encB, encT, mask, attnOut, wOut);
}
